// VectorGate_22677427322904
// MI455X (gfx1250) — matrix operand order repaired and checked by reading, not compiled, not run on hardware
//
#include <hip/hip_runtime.h>
#include <stdint.h>

#define DH 64      // hidden dim
#define DM 128     // mlp dim
#define G3 192     // 3*DH (GRU gates)
#define BN_EPS 1e-5f

typedef __bf16 bf16;
typedef bf16  bf16x8 __attribute__((ext_vector_type(8)));
typedef bf16  v16bf  __attribute__((ext_vector_type(16)));
typedef float v8f    __attribute__((ext_vector_type(8)));
typedef uint32_t u32x4 __attribute__((ext_vector_type(4)));
typedef int   i32x4 __attribute__((ext_vector_type(4)));
typedef int   i32x8 __attribute__((ext_vector_type(8)));

// ---------------- TDM availability (this toolchain: 6-arg form) ------------
#ifndef USE_TDM
# if defined(__has_builtin)
#  if __has_builtin(__builtin_amdgcn_tensor_load_to_lds) && __has_builtin(__builtin_amdgcn_s_wait_tensorcnt)
#   define USE_TDM 1
#  else
#   define USE_TDM 0
#  endif
# else
#  define USE_TDM 0
# endif
#endif

// Stage `count` bf16 elements from global (weight arena) into LDS.
// TDM path: 1-row 2D descriptor (data_size=2B, tile_dim0=count<=16384),
// issued by wave 0, waited with s_wait_tensorcnt 0; caller barriers after.
__device__ __forceinline__ void lds_copy_w(bf16* __restrict__ dst,
                                           const bf16* __restrict__ src,
                                           int count, int tid, int nthreads) {
#if USE_TDM
  if (tid < 32) {
    uint64_t ga = (uint64_t)(uintptr_t)src;
    uint32_t la = (uint32_t)(uintptr_t)dst;     // LDS byte offset (addr[31:0])
    u32x4 g0;
    g0.x = 1u;                                   // count=1 valid descriptor
    g0.y = la;                                   // lds_addr [63:32]
    g0.z = (uint32_t)ga;                         // global_addr low
    g0.w = (uint32_t)((ga >> 32) & 0x1FFFFFFull) | (2u << 30); // addr hi | type=2
    uint32_t dim0 = (uint32_t)count;
    i32x8 g1;
    g1.s0 = (int)(1u << 16);                     // data_size=1 -> 2 bytes
    g1.s1 = (int)((dim0 & 0xFFFFu) << 16);       // tensor_dim0[15:0]
    g1.s2 = (int)(((dim0 >> 16) & 0xFFFFu) | (1u << 16)); // dim0 hi | tensor_dim1=1
    g1.s3 = (int)((dim0 & 0xFFFFu) << 16);       // tile_dim0 = count
    g1.s4 = 1;                                   // tile_dim1=1, tile_dim2=0
    g1.s5 = (int)dim0;                           // tensor_dim0_stride low
    g1.s6 = 0;
    g1.s7 = 0;
    i32x4 g2 = {0, 0, 0, 0};
    i32x4 g3 = {0, 0, 0, 0};
    i32x8 g4 = {0, 0, 0, 0, 0, 0, 0, 0};
    __builtin_amdgcn_tensor_load_to_lds(g0, g1, g2, g3, g4, 0);
    __builtin_amdgcn_s_wait_tensorcnt(0);
  }
#else
  const uint4* s = (const uint4*)src;
  uint4* d = (uint4*)dst;
  int n = count >> 3;
  for (int i = tid; i < n; i += nthreads) d[i] = s[i];
#endif
}

// ---------------- WMMA fragment helpers (wave32, 16x16x32 bf16) ------------
// A (16x32, rows in lanes): lane L holds row m=L&15; halves K = k0+hi*8+{0..7}
// in VGPR0-3 and K = k0+hi*8+16+{0..7} in VGPR4-7  (hi = L>>4).
__device__ __forceinline__ v16bf load_frag_a(const bf16* base, int ld,
                                             int row0, int k0, int lane) {
  int m = lane & 15, hi = lane >> 4;
  const bf16* p = base + (size_t)(row0 + m) * ld + k0 + hi * 8;
  bf16x8 lo = *(const bf16x8*)p;
  bf16x8 h8 = *(const bf16x8*)(p + 16);
  return __builtin_shufflevector(lo, h8, 0,1,2,3,4,5,6,7,8,9,10,11,12,13,14,15);
}
// B (32x16 = W^T slice): weights kept [Mout][K] row-major in LDS, so lane L
// (col n=L&15) needs K = k0+hi*8+{0..7} and k0+hi*8+16+{0..7} from row (col0+n).
__device__ __forceinline__ v16bf load_frag_b(const bf16* w, int ld,
                                             int col0, int k0, int lane) {
  int n = lane & 15, hi = lane >> 4;
  const bf16* p = w + (size_t)(col0 + n) * ld + k0 + hi * 8;
  bf16x8 lo = *(const bf16x8*)p;
  bf16x8 h8 = *(const bf16x8*)(p + 16);
  return __builtin_shufflevector(lo, h8, 0,1,2,3,4,5,6,7,8,9,10,11,12,13,14,15);
}
__device__ __forceinline__ v8f wmma_bf16(v16bf a, v16bf b, v8f c) {
  return __builtin_amdgcn_wmma_f32_16x16x32_bf16(false, a, false, b,
                                                 (short)0, c, false, false);
}
__device__ __forceinline__ float sigmoidf_(float x) {
  return 1.f / (1.f + __expf(-x));
}

// ---------------- small utility kernels ------------------------------------
__global__ void conv_bf16_kernel(const float* __restrict__ s,
                                 bf16* __restrict__ d, int n) {
  int i = blockIdx.x * blockDim.x + threadIdx.x;
  if (i < n) d[i] = (bf16)s[i];
}
__global__ void zero_kernel(float* __restrict__ p, int n) {
  int i = blockIdx.x * blockDim.x + threadIdx.x;
  if (i < n) p[i] = 0.f;
}
__global__ void init_kernel(const float* __restrict__ hs_init,
                            const float* __restrict__ ew,
                            const float* __restrict__ eb,
                            float* __restrict__ hs, float* __restrict__ hf,
                            int total) {
  int i = blockIdx.x * blockDim.x + threadIdx.x;
  if (i < total) {
    hs[i] = hs_init[i];
    int c = i & (DH - 1);
    hf[i] = ew[c] + eb[c];     // Linear(1->D) applied to ones(1): w[:,0]+b
  }
}
__global__ void copyout_kernel(const float* __restrict__ hs,
                               const float* __restrict__ hf,
                               float* __restrict__ out, int nd) {
  int i = blockIdx.x * blockDim.x + threadIdx.x;
  if (i < nd) { out[i] = hs[i]; out[nd + i] = hf[i]; }
}

// ---------------- edge MLP + scatter-add (MlpAggr, aggr='add') -------------
// 64 edges / workgroup, 4 waves. Fused 3-layer MLP K1->128->128->64, bf16 WMMA.
__global__ __launch_bounds__(128) void edge_mlp_kernel(
    const float* __restrict__ featA, const float* __restrict__ featB, int K1,
    const int* __restrict__ src, const int* __restrict__ dst,
    float* __restrict__ msg,
    const bf16* __restrict__ warena, int w1o, int w2o, int w3o,
    const float* __restrict__ b1, const float* __restrict__ b2,
    const float* __restrict__ b3, int nE) {
  __shared__ alignas(16) bf16 sW[DM * DM];   // 32 KB, reused W1->W2->W3
  __shared__ alignas(16) bf16 sA[64 * DM];   // 16 KB activations (ping in place)
  __shared__ int sDst[64];

  const int tid = threadIdx.x;
  const int lane = tid & 31;
  const int wv = tid >> 5;
  const int m16 = lane & 15;
  const int hi = lane >> 4;
  const int e0 = blockIdx.x * 64;
  const v8f vzero = {0.f,0.f,0.f,0.f,0.f,0.f,0.f,0.f};

  // gather X (bf16) : rows = messages feat[src[e]], optionally concat(A|B)
  for (int i = tid; i < 64 * K1; i += 128) {
    int r = i / K1, c = i - r * K1;
    int e = e0 + r;
    int sn = (e < nE) ? src[e] : 0;
    float v = (c < DH) ? featA[(size_t)sn * DH + c]
                       : featB[(size_t)sn * DH + (c - DH)];
    sA[r * K1 + c] = (bf16)v;
  }
  if (tid < 64) {
    int e = e0 + tid;
    sDst[tid] = (e < nE) ? dst[e] : 0;
  }
  __builtin_prefetch(warena + w2o, 0, 1);
  lds_copy_w(sW, warena + w1o, DM * K1, tid, 128);
  __syncthreads();

  // layer 1: [64 x K1] @ W1^T -> 128, relu
  v8f acc1[8];
#pragma unroll
  for (int ct = 0; ct < 8; ++ct) acc1[ct] = vzero;
  for (int k = 0; k < K1; k += 32) {
    v16bf a = load_frag_a(sA, K1, wv * 16, k, lane);
#pragma unroll
    for (int ct = 0; ct < 8; ++ct)
      acc1[ct] = wmma_bf16(a, load_frag_b(sW, K1, ct * 16, k, lane), acc1[ct]);
  }
  __syncthreads();
#pragma unroll
  for (int ct = 0; ct < 8; ++ct) {
    int col = ct * 16 + m16;
    float bias = b1[col];
#pragma unroll
    for (int r = 0; r < 8; ++r) {
      float v = acc1[ct][r] + bias;
      v = v > 0.f ? v : 0.f;
      sA[(wv * 16 + hi * 8 + r) * DM + col] = (bf16)v;
    }
  }
  lds_copy_w(sW, warena + w2o, DM * DM, tid, 128);
  __syncthreads();

  // layer 2: 128 -> 128, relu
  v8f acc2[8];
#pragma unroll
  for (int ct = 0; ct < 8; ++ct) acc2[ct] = vzero;
  for (int k = 0; k < DM; k += 32) {
    v16bf a = load_frag_a(sA, DM, wv * 16, k, lane);
#pragma unroll
    for (int ct = 0; ct < 8; ++ct)
      acc2[ct] = wmma_bf16(a, load_frag_b(sW, DM, ct * 16, k, lane), acc2[ct]);
  }
  __syncthreads();
#pragma unroll
  for (int ct = 0; ct < 8; ++ct) {
    int col = ct * 16 + m16;
    float bias = b2[col];
#pragma unroll
    for (int r = 0; r < 8; ++r) {
      float v = acc2[ct][r] + bias;
      v = v > 0.f ? v : 0.f;
      sA[(wv * 16 + hi * 8 + r) * DM + col] = (bf16)v;
    }
  }
  lds_copy_w(sW, warena + w3o, DH * DM, tid, 128);
  __syncthreads();

  // layer 3: 128 -> 64 (+bias), scatter-add into msg[dst]
  v8f acc3[4];
#pragma unroll
  for (int ct = 0; ct < 4; ++ct) acc3[ct] = vzero;
  for (int k = 0; k < DM; k += 32) {
    v16bf a = load_frag_a(sA, DM, wv * 16, k, lane);
#pragma unroll
    for (int ct = 0; ct < 4; ++ct)
      acc3[ct] = wmma_bf16(a, load_frag_b(sW, DM, ct * 16, k, lane), acc3[ct]);
  }
#pragma unroll
  for (int ct = 0; ct < 4; ++ct) {
    int col = ct * 16 + m16;
    float bias = b3[col];
#pragma unroll
    for (int r = 0; r < 8; ++r) {
      int row = wv * 16 + hi * 8 + r;
      if (e0 + row < nE)
        atomicAdd(&msg[(size_t)sDst[row] * DH + col], acc3[ct][r] + bias);
    }
  }
}

// ---------------- masked single-step GRU (torch semantics) -----------------
// 64 nodes / workgroup. gi = x@W_ih^T+b_ih (WMMA -> LDS bf16),
// gh in regs; r/z/n blocks of a hidden unit live in the same lane.
__global__ __launch_bounds__(128) void gru_kernel(
    const float* __restrict__ xin, float* __restrict__ hvec,
    const int* __restrict__ gate, const int* __restrict__ flev,
    int level, int gtype,
    const bf16* __restrict__ warena, int wih_o, int whh_o,
    const float* __restrict__ b_ih, const float* __restrict__ b_hh, int nN) {
  __shared__ alignas(16) bf16 sW[G3 * DH];    // 24 KB (w_ih then w_hh)
  __shared__ alignas(16) bf16 sXH[64 * DH];   // 8 KB (x then h)
  __shared__ alignas(16) bf16 sGI[64 * G3];   // 24 KB
  __shared__ int sMask[64];
  const int tid = threadIdx.x, lane = tid & 31, wv = tid >> 5;
  const int m16 = lane & 15, hi = lane >> 4;
  const int n0 = blockIdx.x * 64;
  const v8f vzero = {0.f,0.f,0.f,0.f,0.f,0.f,0.f,0.f};

  for (int i = tid; i < 64 * DH; i += 128) {
    int r = i >> 6, c = i & 63;
    int node = n0 + r;
    sXH[i] = (bf16)((node < nN) ? xin[(size_t)node * DH + c] : 0.f);
  }
  if (tid < 64) {
    int node = n0 + tid;
    sMask[tid] = (node < nN) && (flev[node] == level) && (gate[node] == gtype);
  }
  lds_copy_w(sW, warena + wih_o, G3 * DH, tid, 128);
  __syncthreads();

  for (int ct = 0; ct < 12; ++ct) {
    v8f acc = vzero;
    for (int k = 0; k < DH; k += 32) {
      v16bf a = load_frag_a(sXH, DH, wv * 16, k, lane);
      acc = wmma_bf16(a, load_frag_b(sW, DH, ct * 16, k, lane), acc);
    }
    int col = ct * 16 + m16;
    float bias = b_ih[col];
#pragma unroll
    for (int r = 0; r < 8; ++r)
      sGI[(wv * 16 + hi * 8 + r) * G3 + col] = (bf16)(acc[r] + bias);
  }
  __syncthreads();

  for (int i = tid; i < 64 * DH; i += 128) {
    int r = i >> 6, c = i & 63;
    int node = n0 + r;
    sXH[i] = (bf16)((node < nN) ? hvec[(size_t)node * DH + c] : 0.f);
  }
  lds_copy_w(sW, warena + whh_o, G3 * DH, tid, 128);
  __syncthreads();

  for (int ctg = 0; ctg < 4; ++ctg) {
    v8f aR = vzero, aZ = vzero, aN = vzero;
    for (int k = 0; k < DH; k += 32) {
      v16bf a = load_frag_a(sXH, DH, wv * 16, k, lane);
      aR = wmma_bf16(a, load_frag_b(sW, DH, ctg * 16, k, lane), aR);
      aZ = wmma_bf16(a, load_frag_b(sW, DH, (ctg + 4) * 16, k, lane), aZ);
      aN = wmma_bf16(a, load_frag_b(sW, DH, (ctg + 8) * 16, k, lane), aN);
    }
    int d = ctg * 16 + m16;
    float bR = b_hh[d], bZ = b_hh[DH + d], bN = b_hh[2 * DH + d];
#pragma unroll
    for (int r = 0; r < 8; ++r) {
      int row = wv * 16 + hi * 8 + r;
      float giR = (float)sGI[row * G3 + d];
      float giZ = (float)sGI[row * G3 + DH + d];
      float giN = (float)sGI[row * G3 + 2 * DH + d];
      float rg = sigmoidf_(giR + aR[r] + bR);
      float zg = sigmoidf_(giZ + aZ[r] + bZ);
      float ng = tanhf(giN + rg * (aN[r] + bN));
      float hold = (float)sXH[row * DH + d];
      float hnew = (1.f - zg) * ng + zg * hold;
      if (sMask[row]) hvec[(size_t)(n0 + row) * DH + d] = hnew;
    }
  }
}

// ---------------- readout: Linear+BN+ReLU x2, Linear(128->1) ---------------
__global__ __launch_bounds__(128) void readout_kernel(
    const float* __restrict__ hf, float* __restrict__ prob,
    const bf16* __restrict__ warena, int w1o, int w2o,
    const float* __restrict__ b1, const float* __restrict__ b2,
    const float* __restrict__ w3, const float* __restrict__ b3,
    const float* __restrict__ g0, const float* __restrict__ bb0,
    const float* __restrict__ rm0, const float* __restrict__ rv0,
    const float* __restrict__ g1p, const float* __restrict__ bb1,
    const float* __restrict__ rm1, const float* __restrict__ rv1, int nN) {
  __shared__ alignas(16) bf16 sW[DM * DM];
  __shared__ alignas(16) bf16 sA[64 * DM];
  __shared__ alignas(16) bf16 sX[64 * DH];
  const int tid = threadIdx.x, lane = tid & 31, wv = tid >> 5;
  const int m16 = lane & 15, hi = lane >> 4;
  const int n0 = blockIdx.x * 64;
  const v8f vzero = {0.f,0.f,0.f,0.f,0.f,0.f,0.f,0.f};

  for (int i = tid; i < 64 * DH; i += 128) {
    int r = i >> 6, c = i & 63;
    int node = n0 + r;
    sX[i] = (bf16)((node < nN) ? hf[(size_t)node * DH + c] : 0.f);
  }
  lds_copy_w(sW, warena + w1o, DM * DH, tid, 128);
  __syncthreads();

  v8f acc[8];
#pragma unroll
  for (int ct = 0; ct < 8; ++ct) acc[ct] = vzero;
  for (int k = 0; k < DH; k += 32) {
    v16bf a = load_frag_a(sX, DH, wv * 16, k, lane);
#pragma unroll
    for (int ct = 0; ct < 8; ++ct)
      acc[ct] = wmma_bf16(a, load_frag_b(sW, DH, ct * 16, k, lane), acc[ct]);
  }
#pragma unroll
  for (int ct = 0; ct < 8; ++ct) {
    int col = ct * 16 + m16;
    float sc = g0[col] * rsqrtf(rv0[col] + BN_EPS);
    float sh = bb0[col] - rm0[col] * sc;
#pragma unroll
    for (int r = 0; r < 8; ++r) {
      float v = (acc[ct][r] + b1[col]) * sc + sh;
      v = v > 0.f ? v : 0.f;
      sA[(wv * 16 + hi * 8 + r) * DM + col] = (bf16)v;
    }
  }
  __syncthreads();
  lds_copy_w(sW, warena + w2o, DM * DM, tid, 128);
  __syncthreads();

  v8f acc2[8];
#pragma unroll
  for (int ct = 0; ct < 8; ++ct) acc2[ct] = vzero;
  for (int k = 0; k < DM; k += 32) {
    v16bf a = load_frag_a(sA, DM, wv * 16, k, lane);
#pragma unroll
    for (int ct = 0; ct < 8; ++ct)
      acc2[ct] = wmma_bf16(a, load_frag_b(sW, DM, ct * 16, k, lane), acc2[ct]);
  }
  __syncthreads();
#pragma unroll
  for (int ct = 0; ct < 8; ++ct) {
    int col = ct * 16 + m16;
    float sc = g1p[col] * rsqrtf(rv1[col] + BN_EPS);
    float sh = bb1[col] - rm1[col] * sc;
#pragma unroll
    for (int r = 0; r < 8; ++r) {
      float v = (acc2[ct][r] + b2[col]) * sc + sh;
      v = v > 0.f ? v : 0.f;
      sA[(wv * 16 + hi * 8 + r) * DM + col] = (bf16)v;
    }
  }
  __syncthreads();
  if (tid < 64) {
    int node = n0 + tid;
    if (node < nN) {
      float s = 0.f;
      for (int c = 0; c < DM; ++c) s += (float)sA[tid * DM + c] * w3[c];
      prob[node] = s + b3[0];
    }
  }
}

// ---------------- host orchestration ---------------------------------------
extern "C" void kernel_launch(void* const* d_in, const int* in_sizes, int n_in,
                              void* d_out, int out_size, void* d_ws,
                              size_t ws_size, hipStream_t stream) {
  (void)out_size; (void)ws_size;
  const int nN = in_sizes[1];
  const int nE = in_sizes[0] / 2;
  const int* edge = (const int*)d_in[0];
  const int* src = edge;
  const int* dst = edge + nE;
  const int* gate = (const int*)d_in[1];
  const int* flev = (const int*)d_in[2];
  const float* hs_init = (const float*)d_in[3];
  const float* P[64] = {};
  for (int i = 4; i < n_in && i < 64; ++i) P[i] = (const float*)d_in[i];

  const size_t nd = (size_t)nN * DH;
  float* hs   = (float*)d_ws;
  float* hf   = hs + nd;
  float* msg1 = hf + nd;
  float* msg2 = msg1 + nd;
  bf16*  wbf  = (bf16*)(msg2 + nd);

  // bf16 weight arena offsets (elements)
  enum {
    AS_W1 = 0,                 AS_W2 = AS_W1 + DM*DH,  AS_W3 = AS_W2 + DM*DM,
    AF_W1 = AS_W3 + DH*DM,     AF_W2 = AF_W1 + DM*DM,  AF_W3 = AF_W2 + DM*DM,
    NS_W1 = AF_W3 + DH*DM,     NS_W2 = NS_W1 + DM*DH,  NS_W3 = NS_W2 + DM*DM,
    NF_W1 = NS_W3 + DH*DM,     NF_W2 = NF_W1 + DM*DH,  NF_W3 = NF_W2 + DM*DM,
    GAS_IH = NF_W3 + DH*DM,    GAS_HH = GAS_IH + G3*DH,
    GAF_IH = GAS_HH + G3*DH,   GAF_HH = GAF_IH + G3*DH,
    GNS_IH = GAF_HH + G3*DH,   GNS_HH = GNS_IH + G3*DH,
    GNF_IH = GNS_HH + G3*DH,   GNF_HH = GNF_IH + G3*DH,
    RO_W1 = GNF_HH + G3*DH,    RO_W2 = RO_W1 + DM*DH
  };

  auto conv = [&](int pidx, int off, int count) {
    conv_bf16_kernel<<<(count + 255) / 256, 256, 0, stream>>>(P[pidx], wbf + off, count);
  };
  conv(4,  AS_W1, DM*DH); conv(6,  AS_W2, DM*DM); conv(8,  AS_W3, DH*DM);
  conv(10, AF_W1, DM*DM); conv(12, AF_W2, DM*DM); conv(14, AF_W3, DH*DM);
  conv(16, NS_W1, DM*DH); conv(18, NS_W2, DM*DM); conv(20, NS_W3, DH*DM);
  conv(22, NF_W1, DM*DH); conv(24, NF_W2, DM*DM); conv(26, NF_W3, DH*DM);
  conv(28, GAS_IH, G3*DH); conv(29, GAS_HH, G3*DH);
  conv(32, GAF_IH, G3*DH); conv(33, GAF_HH, G3*DH);
  conv(36, GNS_IH, G3*DH); conv(37, GNS_HH, G3*DH);
  conv(40, GNF_IH, G3*DH); conv(41, GNF_HH, G3*DH);
  conv(46, RO_W1, DM*DH);  conv(48, RO_W2, DM*DM);

  init_kernel<<<((int)nd + 255) / 256, 256, 0, stream>>>(hs_init, P[44], P[45],
                                                         hs, hf, (int)nd);
  const int egrid = (nE + 63) / 64;
  const int ngrid = (nN + 63) / 64;

  for (int level = 1; level < 4; ++level) {
    // AND step: both aggregations read pre-update hs/hf (== torch in-place order)
    zero_kernel<<<((int)(2*nd) + 255) / 256, 256, 0, stream>>>(msg1, (int)(2*nd));
    edge_mlp_kernel<<<egrid, 128, 0, stream>>>(hs, hs, DH, src, dst, msg1, wbf,
        AS_W1, AS_W2, AS_W3, P[5], P[7], P[9], nE);
    edge_mlp_kernel<<<egrid, 128, 0, stream>>>(hs, hf, 2*DH, src, dst, msg2, wbf,
        AF_W1, AF_W2, AF_W3, P[11], P[13], P[15], nE);
    gru_kernel<<<ngrid, 128, 0, stream>>>(msg1, hs, gate, flev, level, 1, wbf,
        GAS_IH, GAS_HH, P[30], P[31], nN);
    gru_kernel<<<ngrid, 128, 0, stream>>>(msg2, hf, gate, flev, level, 1, wbf,
        GAF_IH, GAF_HH, P[34], P[35], nN);
    // NOT step: aggregations read post-AND hs/hf
    zero_kernel<<<((int)(2*nd) + 255) / 256, 256, 0, stream>>>(msg1, (int)(2*nd));
    edge_mlp_kernel<<<egrid, 128, 0, stream>>>(hs, hs, DH, src, dst, msg1, wbf,
        NS_W1, NS_W2, NS_W3, P[17], P[19], P[21], nE);
    edge_mlp_kernel<<<egrid, 128, 0, stream>>>(hf, hf, DH, src, dst, msg2, wbf,
        NF_W1, NF_W2, NF_W3, P[23], P[25], P[27], nE);
    gru_kernel<<<ngrid, 128, 0, stream>>>(msg1, hs, gate, flev, level, 2, wbf,
        GNS_IH, GNS_HH, P[38], P[39], nN);
    gru_kernel<<<ngrid, 128, 0, stream>>>(msg2, hf, gate, flev, level, 2, wbf,
        GNF_IH, GNF_HH, P[42], P[43], nN);
  }

  float* out = (float*)d_out;
  copyout_kernel<<<((int)nd + 255) / 256, 256, 0, stream>>>(hs, hf, out, (int)nd);
  readout_kernel<<<ngrid, 128, 0, stream>>>(hf, out + 2 * nd, wbf, RO_W1, RO_W2,
      P[47], P[49], P[50], P[51],
      P[52], P[53], P[54], P[55],
      P[56], P[57], P[58], P[59], nN);
}
